// BinderQuantization_11897059410185
// MI455X (gfx1250) — compile-verified, matrix operand order checked by reading, not run on hardware
//
#include <hip/hip_runtime.h>
#include <stdint.h>

// ---------------------------------------------------------------------------
// Types for CDNA5 WMMA
// ---------------------------------------------------------------------------
typedef __attribute__((ext_vector_type(16))) __bf16 v16bf;
typedef __attribute__((ext_vector_type(8)))  float  v8f;

struct alignas(16) U128 { uint32_t x[4]; };
union FragU { v16bf v; U128 q[2]; };

#define BM 128
#define BN 128
#define BKK 32
#define USE_ASYNC 1

__device__ __forceinline__ unsigned short f2bf(float f) {
    union { float f; uint32_t u; } v; v.f = f;
    uint32_t u = v.u;
    uint32_t r = u + 0x7FFFu + ((u >> 16) & 1u);   // round-to-nearest-even
    return (unsigned short)(r >> 16);
}

// CDNA5 async global -> LDS copy (ASYNCcnt-tracked). Generic LDS pointers carry
// the LDS byte address in their low 32 bits (aperture rule, ISA 10.2).
__device__ __forceinline__ void async_copy_b128(const void* g, void* l) {
#if USE_ASYNC
    asm volatile("global_load_async_to_lds_b128 %0, %1, off"
                 :: "v"((uint32_t)(uintptr_t)l), "v"((uint64_t)(uintptr_t)g)
                 : "memory");
#else
    *(U128*)l = *(const U128*)g;
#endif
}

__device__ __forceinline__ void async_wait0() {
#if USE_ASYNC
#if __has_builtin(__builtin_amdgcn_s_wait_asynccnt)
    __builtin_amdgcn_s_wait_asynccnt(0);
#else
    asm volatile("s_wait_asynccnt 0x0" ::: "memory");
#endif
#endif
}

// ---------------------------------------------------------------------------
// fp32 -> bf16 conversion (no transpose)
// ---------------------------------------------------------------------------
__global__ void f2bf_kernel(const float* __restrict__ in,
                            unsigned short* __restrict__ out, int n) {
    int i = blockIdx.x * blockDim.x + threadIdx.x;
    if (i < n) out[i] = f2bf(in[i]);
}

// fp32 (K x N) -> bf16 transposed (N x K)
__global__ void wT_kernel(const float* __restrict__ w,
                          unsigned short* __restrict__ wT, int K, int N) {
    int idx = blockIdx.x * blockDim.x + threadIdx.x;
    if (idx < K * N) {
        int k = idx / N, n = idx % N;          // coalesced read over n
        wT[(size_t)n * K + k] = f2bf(w[idx]);
    }
}

// ---------------------------------------------------------------------------
// Tiled bf16 WMMA GEMM:  C[M,N] = act(A[M,K] * B^T + bias)
//   A element (m,k) at A[m*lda + k]
//   B element (k,n) at B[n*ldb + k]      (always N x K row-major)
//   blockIdx.z batches independent GEMMs via aZ/bZ/cZ element strides.
// Double-buffered LDS staged with GLOBAL_LOAD_ASYNC_TO_LDS_B128.
// ---------------------------------------------------------------------------
template <bool RELU, bool OUT_BF16>
__global__ __launch_bounds__(256)
void gemm_bf16_kernel(const unsigned short* __restrict__ A, int lda, size_t aZ,
                      const unsigned short* __restrict__ B, int ldb, size_t bZ,
                      const float* __restrict__ bias,
                      float* __restrict__ Cf, unsigned short* __restrict__ Cb,
                      int ldc, size_t cZ, int K) {
    __shared__ alignas(16) unsigned short sA[2][BM][BKK];
    __shared__ alignas(16) unsigned short sB[2][BN][BKK];

    A += (size_t)blockIdx.z * aZ;
    B += (size_t)blockIdx.z * bZ;
    if (OUT_BF16) Cb += (size_t)blockIdx.z * cZ;
    else          Cf += (size_t)blockIdx.z * cZ;

    const int tid  = threadIdx.x;
    const int lane = tid & 31;
    const int wid  = tid >> 5;        // 0..7
    const int wm   = wid & 3;         // 4 waves along M (32 rows each)
    const int wn   = wid >> 2;        // 2 waves along N (64 cols each)
    const int mBlock = blockIdx.y * BM;
    const int nBlock = blockIdx.x * BN;

    v8f acc[2][4];
    const v8f vzero = {0.f, 0.f, 0.f, 0.f, 0.f, 0.f, 0.f, 0.f};
#pragma unroll
    for (int i = 0; i < 2; ++i)
#pragma unroll
        for (int j = 0; j < 4; ++j) acc[i][j] = vzero;

    const int nk = K / BKK;

    // Stage one 128x32 A tile + 128x32 B tile. Each tile = 512 16B chunks,
    // 2 per thread -> 4 async b128 copies per thread per K-step.
    auto loadTile = [&](int buf, int kt) {
        const int kbase = kt * BKK;
#pragma unroll
        for (int a = 0; a < 2; ++a) {
            int c = tid + a * 256;
            int r = c >> 2, h = (c & 3) * 8;
            async_copy_b128(&A[(size_t)(mBlock + r) * lda + kbase + h],
                            &sA[buf][r][h]);
        }
#pragma unroll
        for (int a = 0; a < 2; ++a) {
            int c = tid + a * 256;
            int r = c >> 2, h = (c & 3) * 8;
            async_copy_b128(&B[(size_t)(nBlock + r) * ldb + kbase + h],
                            &sB[buf][r][h]);
        }
    };

    loadTile(0, 0);
    async_wait0();
    __syncthreads();

    const int l16 = lane & 15;
    const int kb  = (lane >> 4) * 8;   // 16-bit fragment layout: lanes 16-31 hold K+8

    for (int kt = 0; kt < nk; ++kt) {
        const int buf = kt & 1;
        if (kt + 1 < nk) loadTile(buf ^ 1, kt + 1);

        FragU af[2];
#pragma unroll
        for (int i = 0; i < 2; ++i) {
            int m = wm * 32 + i * 16 + l16;
            af[i].q[0] = *(const U128*)&sA[buf][m][kb];
            af[i].q[1] = *(const U128*)&sA[buf][m][kb + 16];
        }
        FragU bfr[4];
#pragma unroll
        for (int j = 0; j < 4; ++j) {
            int n = wn * 64 + j * 16 + l16;
            bfr[j].q[0] = *(const U128*)&sB[buf][n][kb];
            bfr[j].q[1] = *(const U128*)&sB[buf][n][kb + 16];
        }
#pragma unroll
        for (int i = 0; i < 2; ++i)
#pragma unroll
            for (int j = 0; j < 4; ++j)
                acc[i][j] = __builtin_amdgcn_wmma_f32_16x16x32_bf16(
                    false, af[i].v, false, bfr[j].v, (short)0, acc[i][j],
                    false, false);

        async_wait0();     // next tile resident before anyone signals
        __syncthreads();
    }

    // Epilogue. C/D layout: VGPR r -> M = r (lanes 0-15), M = 8+r (lanes 16-31).
    const int hi8 = (lane >> 4) * 8;
#pragma unroll
    for (int i = 0; i < 2; ++i) {
#pragma unroll
        for (int j = 0; j < 4; ++j) {
            int n = nBlock + wn * 64 + j * 16 + l16;
            float bv = bias ? bias[n] : 0.f;
#pragma unroll
            for (int r = 0; r < 8; ++r) {
                int m = mBlock + wm * 32 + i * 16 + hi8 + r;
                float v = acc[i][j][r] + bv;
                if (RELU) v = v > 0.f ? v : 0.f;
                if (OUT_BF16) Cb[(size_t)m * ldc + n] = f2bf(v);
                else          Cf[(size_t)m * ldc + n] = v;
            }
        }
    }
}

// ---------------------------------------------------------------------------
// LayerNorm of mem rows (V*T x 256): emits row-major bf16 and e-major bf16.
// Row index = v*T + t.  outT[((t*256)+e)*V + v]
// ---------------------------------------------------------------------------
__global__ __launch_bounds__(256)
void ln_mem_kernel(const float* __restrict__ in,
                   unsigned short* __restrict__ outR,
                   unsigned short* __restrict__ outT) {
    __shared__ float red[256];
    const int tid = threadIdx.x;
    const int row = blockIdx.x;
    const size_t base = (size_t)row * 256;
    float x = in[base + tid];

    red[tid] = x; __syncthreads();
    for (int s = 128; s > 0; s >>= 1) {
        if (tid < s) red[tid] += red[tid + s];
        __syncthreads();
    }
    float mean = red[0] / 256.f; __syncthreads();

    float dx = x - mean;
    red[tid] = dx * dx; __syncthreads();
    for (int s = 128; s > 0; s >>= 1) {
        if (tid < s) red[tid] += red[tid + s];
        __syncthreads();
    }
    float var = red[0] / 256.f;
    float y = dx * rsqrtf(var + 1e-5f);
    unsigned short b = f2bf(y);
    outR[base + tid] = b;
    int v = row >> 2, t = row & 3;
    outT[((size_t)t * 256 + tid) * 8192 + v] = b;
}

// q = layernorm(z) * E^-0.5, written into per-t compacted (t, b*K+k, e) bf16
__global__ __launch_bounds__(256)
void ln_q_kernel(const float* __restrict__ z, unsigned short* __restrict__ q16) {
    __shared__ float red[256];
    const int tid = threadIdx.x;
    const int row = blockIdx.x;              // (b*K+k)*T + t
    float x = z[(size_t)row * 256 + tid];

    red[tid] = x; __syncthreads();
    for (int s = 128; s > 0; s >>= 1) {
        if (tid < s) red[tid] += red[tid + s];
        __syncthreads();
    }
    float mean = red[0] / 256.f; __syncthreads();

    float dx = x - mean;
    red[tid] = dx * dx; __syncthreads();
    for (int s = 128; s > 0; s >>= 1) {
        if (tid < s) red[tid] += red[tid + s];
        __syncthreads();
    }
    float var = red[0] / 256.f;
    float y = dx * rsqrtf(var + 1e-5f) * 0.0625f;   // E^-0.5 = 1/16

    int t = row & 3, m = row >> 2;
    q16[((size_t)t * 512 + m) * 256 + tid] = f2bf(y);
}

// ---------------------------------------------------------------------------
// Fused softmax (over V=8192) + first-occurrence argmax (tokens).
// Row g = t*512 + m of S; P gets bf16 probabilities; tokens[(m*T)+t] = idx.
// ---------------------------------------------------------------------------
__global__ __launch_bounds__(256)
void softmax_argmax_kernel(const float* __restrict__ S,
                           unsigned short* __restrict__ P,
                           float* __restrict__ tokens) {
    __shared__ float sval[256];
    __shared__ int   sidx[256];
    const int tid = threadIdx.x;
    const int g   = blockIdx.x;
    const float* row = S + (size_t)g * 8192;

    float mx = -3.402823466e38f;
    int   ai = 0;
    for (int i = tid; i < 8192; i += 256) {
        float v = row[i];
        if (v > mx) { mx = v; ai = i; }
    }
    sval[tid] = mx; sidx[tid] = ai; __syncthreads();
    for (int s = 128; s > 0; s >>= 1) {
        if (tid < s) {
            float vo = sval[tid + s]; int io = sidx[tid + s];
            if (vo > sval[tid] || (vo == sval[tid] && io < sidx[tid])) {
                sval[tid] = vo; sidx[tid] = io;
            }
        }
        __syncthreads();
    }
    mx = sval[0];
    const int best = sidx[0];
    __syncthreads();

    float sum = 0.f;
    for (int i = tid; i < 8192; i += 256) sum += expf(row[i] - mx);
    sval[tid] = sum; __syncthreads();
    for (int s = 128; s > 0; s >>= 1) {
        if (tid < s) sval[tid] += sval[tid + s];
        __syncthreads();
    }
    float inv = 1.f / sval[0];

    unsigned short* prow = P + (size_t)g * 8192;
    for (int i = tid; i < 8192; i += 256)
        prow[i] = f2bf(expf(row[i] - mx) * inv);

    if (tid == 0) {
        int t = g >> 9, m = g & 511;
        tokens[(size_t)m * 4 + t] = (float)best;
    }
}

// ---------------------------------------------------------------------------
// Host launcher
// ---------------------------------------------------------------------------
extern "C" void kernel_launch(void* const* d_in, const int* in_sizes, int n_in,
                              void* d_out, int out_size, void* d_ws,
                              size_t ws_size, hipStream_t stream) {
    (void)in_sizes; (void)n_in; (void)out_size; (void)ws_size;

    const float* z   = (const float*)d_in[0];
    const float* emb = (const float*)d_in[1];
    const float* w1  = (const float*)d_in[2];
    const float* b1  = (const float*)d_in[3];
    const float* w2  = (const float*)d_in[4];
    const float* b2  = (const float*)d_in[5];
    const float* w3  = (const float*)d_in[6];
    const float* b3  = (const float*)d_in[7];
    const float* w4  = (const float*)d_in[8];
    const float* b4  = (const float*)d_in[9];

    constexpr int T = 4, E = 256, V = 8192, H = 1024;
    constexpr int Mmem = V * T;        // 32768
    constexpr int Mq   = 512;          // b*K rows per t

    char* ws = (char*)d_ws;
    size_t off = 0;
    auto bump = [&](size_t bytes) {
        size_t o = off;
        off += (bytes + 255) & ~(size_t)255;
        return o;
    };

    unsigned short* emb16 = (unsigned short*)(ws + bump((size_t)Mmem * E * 2));
    unsigned short* w1T   = (unsigned short*)(ws + bump((size_t)E * H * 2));
    unsigned short* w2T   = (unsigned short*)(ws + bump((size_t)H * H * 2));
    unsigned short* w3T   = (unsigned short*)(ws + bump((size_t)H * H * 2));
    unsigned short* w4T   = (unsigned short*)(ws + bump((size_t)H * E * 2));
    unsigned short* q16   = (unsigned short*)(ws + bump((size_t)T * Mq * E * 2));
    char* bufA = ws + bump((size_t)Mmem * H * 2);     // 64 MB
    char* bufB = ws + bump((size_t)Mmem * H * 2);     // 64 MB
    unsigned short* P = (unsigned short*)(ws + bump((size_t)T * Mq * V * 2)); // 32 MB

    unsigned short* h1 = (unsigned short*)bufA;
    unsigned short* h2 = (unsigned short*)bufB;
    unsigned short* h3 = (unsigned short*)bufA;                   // h1 dead
    float* mem_raw = (float*)bufB;                                // h2 dead
    unsigned short* mem_ln16 = (unsigned short*)bufA;             // h3 dead (16 MB)
    unsigned short* mem_T16 =
        (unsigned short*)(bufA + (size_t)Mmem * E * 2);           // 16 MB
    float* S = (float*)bufB;                                      // mem_raw dead

    float* tokens = (float*)d_out;
    float* zq     = (float*)d_out + 2048;

    // --- stage 0: bf16 conversions / weight transposes ---
    f2bf_kernel<<<Mmem * E / 256, 256, 0, stream>>>(emb, emb16, Mmem * E);
    wT_kernel<<<E * H / 256, 256, 0, stream>>>(w1, w1T, E, H);
    wT_kernel<<<H * H / 256, 256, 0, stream>>>(w2, w2T, H, H);
    wT_kernel<<<H * H / 256, 256, 0, stream>>>(w3, w3T, H, H);
    wT_kernel<<<H * E / 256, 256, 0, stream>>>(w4, w4T, H, E);

    // --- stage 1: codebook MLP (B always N x K row-major) ---
    dim3 gH(H / BN, Mmem / BM, 1);     // (8, 256)
    gemm_bf16_kernel<true, true><<<gH, 256, 0, stream>>>(
        emb16, E, 0, w1T, E, 0, b1, nullptr, h1, H, 0, E);
    gemm_bf16_kernel<true, true><<<gH, 256, 0, stream>>>(
        h1, H, 0, w2T, H, 0, b2, nullptr, h2, H, 0, H);
    gemm_bf16_kernel<true, true><<<gH, 256, 0, stream>>>(
        h2, H, 0, w3T, H, 0, b3, nullptr, h3, H, 0, H);
    dim3 gE(E / BN, Mmem / BM, 1);     // (2, 256)
    gemm_bf16_kernel<false, false><<<gE, 256, 0, stream>>>(
        h3, H, 0, w4T, H, 0, b4, mem_raw, nullptr, E, 0, H);

    // --- stage 2: layernorms ---
    ln_mem_kernel<<<Mmem, 256, 0, stream>>>(mem_raw, mem_ln16, mem_T16);
    ln_q_kernel<<<2048, 256, 0, stream>>>(z, q16);

    // --- stage 3: attention logits  S_t = q_t (512x256) @ mem_t^T (256x8192) ---
    dim3 gAtt(V / BN, Mq / BM, T);     // (64, 4, 4)
    gemm_bf16_kernel<false, false><<<gAtt, 256, 0, stream>>>(
        q16, E, (size_t)Mq * E,
        mem_ln16, T * E, (size_t)E,
        nullptr, S, nullptr, V, (size_t)Mq * V, E);

    // --- stage 4: softmax + argmax (tokens) ---
    softmax_argmax_kernel<<<2048, 256, 0, stream>>>(S, P, tokens);

    // --- stage 5: z_q = P_t (512x8192) @ mem_t (8192x256) via e-major mem^T ---
    dim3 gOut(E / BN, Mq / BM, T);     // (2, 4, 4)
    gemm_bf16_kernel<false, false><<<gOut, 256, 0, stream>>>(
        P, V, (size_t)Mq * V,
        mem_T16, V, (size_t)E * V,
        nullptr, zq, nullptr, T * E, (size_t)E, V);
}
